// GraphAttBn_82042465288994
// MI455X (gfx1250) — matrix operand order repaired and checked by reading, not compiled, not run on hardware
//
#include <hip/hip_runtime.h>
#include <hip/hip_bf16.h>
#include <stdint.h>

typedef __attribute__((ext_vector_type(16))) __bf16 v16bf;
typedef __attribute__((ext_vector_type(8)))  float  v8f;

#define WLDS_STRIDE 132   // 128 + pad to break LDS bank alignment

__device__ __forceinline__ float lrelu(float e) { return e > 0.f ? e : 0.2f * e; }

// monotone float->uint key (orders like float over full range)
__device__ __forceinline__ unsigned fkey(float f) {
  unsigned u = __float_as_uint(f);
  return (u & 0x80000000u) ? ~u : (u | 0x80000000u);
}
__device__ __forceinline__ float unkey(unsigned k) {
  unsigned u = (k & 0x80000000u) ? (k & 0x7fffffffu) : ~k;
  return __uint_as_float(u);
}

// ---------------------------------------------------------------- zero init
__global__ void k_zero(float* __restrict__ p, long n) {
  long i = (long)blockIdx.x * blockDim.x + threadIdx.x;
  if (i < n) p[i] = 0.f;
}

// ------------------------------------------------- h = x @ W  (bf16 WMMA)
// block = 256 threads (8 waves); each wave computes a 16x128 tile of h.
__global__ __launch_bounds__(256)
void k_gemm(const float* __restrict__ x, const float* __restrict__ W,
            float* __restrict__ h, int n_nodes) {
  __shared__ __bf16 wlds[128 * WLDS_STRIDE];
  for (int i = threadIdx.x; i < 128 * 128; i += 256) {
    int r = i >> 7, c = i & 127;
    wlds[r * WLDS_STRIDE + c] = (__bf16)W[i];
  }
  __syncthreads();

  const int wave  = threadIdx.x >> 5;
  const int lane  = threadIdx.x & 31;
  const int lhalf = lane >> 4;        // 0 or 1
  const int l15   = lane & 15;
  const long m0   = (long)blockIdx.x * 128 + wave * 16;
  long rA = m0 + l15;
  if (rA > n_nodes - 1) rA = n_nodes - 1;   // clamp; EXEC must stay all-ones

  v8f acc[8];
#pragma unroll
  for (int t = 0; t < 8; ++t) {
#pragma unroll
    for (int e = 0; e < 8; ++e) acc[t][e] = 0.f;
  }

#pragma unroll
  for (int ks = 0; ks < 4; ++ks) {
    // A fragment (16x32 bf16): lane=M, element e -> K = (e<8 ? e : e+8) + lhalf*8
    const float* xr = x + (size_t)rA * 128 + ks * 32 + lhalf * 8;
    v16bf afrag;
#pragma unroll
    for (int e = 0; e < 8; ++e)  afrag[e] = (__bf16)xr[e];
#pragma unroll
    for (int e = 8; e < 16; ++e) afrag[e] = (__bf16)xr[e + 8];

#pragma unroll
    for (int nt = 0; nt < 8; ++nt) {
      // B fragment (32x16 bf16): lane=N, element e -> K = (e<8 ? e : e+8) + lhalf*8
      const __bf16* wr = &wlds[(ks * 32 + lhalf * 8) * WLDS_STRIDE + nt * 16 + l15];
      v16bf bfrag;
#pragma unroll
      for (int e = 0; e < 16; ++e) bfrag[e] = wr[(e < 8 ? e : e + 8) * WLDS_STRIDE];
      acc[nt] = __builtin_amdgcn_wmma_f32_16x16x32_bf16(
          false, afrag, false, bfrag, (short)0, acc[nt], false, false);
    }
  }

  // C/D layout: VGPR r, lanes 0-15 -> M=r, lanes 16-31 -> M=r+8; N = lane&15
#pragma unroll
  for (int nt = 0; nt < 8; ++nt) {
#pragma unroll
    for (int r = 0; r < 8; ++r) {
      long row = m0 + lhalf * 8 + r;
      if (row < n_nodes) h[row * 128 + nt * 16 + l15] = acc[nt][r];
    }
  }
}

// --------------------------------- per-node attention scalars + init m/den
__global__ void k_node(const float* __restrict__ h,
                       const float* __restrict__ att_src,
                       const float* __restrict__ att_dst,
                       float* __restrict__ a_s, float* __restrict__ a_d,
                       unsigned* __restrict__ mkey, float* __restrict__ den,
                       int n) {
  int i = blockIdx.x * blockDim.x + threadIdx.x;
  if (i >= n * 4) return;
  int node = i >> 2, hh = i & 3;
  const float* hr = h + (size_t)node * 128 + hh * 32;
  const float* as = att_src + hh * 32;
  const float* ad = att_dst + hh * 32;
  float s = 0.f, d = 0.f;
#pragma unroll
  for (int c = 0; c < 32; ++c) { float v = hr[c]; s += v * as[c]; d += v * ad[c]; }
  a_s[i] = s; a_d[i] = d;
  mkey[i] = 0u;        // key-space -infinity
  den[i]  = 0.f;
}

// -------------------------------------------------- segment max over edges
__global__ void k_edge_max(const long long* __restrict__ ei,
                           const float* __restrict__ a_s,
                           const float* __restrict__ a_d,
                           unsigned* __restrict__ mkey, int E_, int EN) {
  int i = blockIdx.x * blockDim.x + threadIdx.x;
  if (i >= EN) return;
  int s, d;
  if (i < E_) { s = (int)ei[i]; d = (int)ei[E_ + i]; } else { s = d = i - E_; }
  const float4 av = *(const float4*)(a_s + (size_t)s * 4);
  const float4 dv = *(const float4*)(a_d + (size_t)d * 4);
  float ev[4] = {av.x + dv.x, av.y + dv.y, av.z + dv.z, av.w + dv.w};
#pragma unroll
  for (int hh = 0; hh < 4; ++hh) {
    unsigned k = fkey(lrelu(ev[hh]));
    __hip_atomic_fetch_max(&mkey[(size_t)d * 4 + hh], k,
                           __ATOMIC_RELAXED, __HIP_MEMORY_SCOPE_AGENT);
  }
}

// keys -> float max values (in place, bitwise)
__global__ void k_mconv(unsigned* __restrict__ m, int n4) {
  int i = blockIdx.x * blockDim.x + threadIdx.x;
  if (i < n4) m[i] = __float_as_uint(unkey(m[i]));
}

// ------------------------------------------------ segment sum of exp(e - m)
__global__ void k_edge_sum(const long long* __restrict__ ei,
                           const float* __restrict__ a_s,
                           const float* __restrict__ a_d,
                           const float* __restrict__ mval,
                           float* __restrict__ den, int E_, int EN) {
  int i = blockIdx.x * blockDim.x + threadIdx.x;
  if (i >= EN) return;
  int s, d;
  if (i < E_) { s = (int)ei[i]; d = (int)ei[E_ + i]; } else { s = d = i - E_; }
  const float4 av = *(const float4*)(a_s + (size_t)s * 4);
  const float4 dv = *(const float4*)(a_d + (size_t)d * 4);
  const float4 mv = *(const float4*)(mval + (size_t)d * 4);
  float ev[4] = {av.x + dv.x, av.y + dv.y, av.z + dv.z, av.w + dv.w};
  float mm[4] = {mv.x, mv.y, mv.z, mv.w};
#pragma unroll
  for (int hh = 0; hh < 4; ++hh) {
    float ex = __expf(lrelu(ev[hh]) - mm[hh]);
    __hip_atomic_fetch_add(&den[(size_t)d * 4 + hh], ex,
                           __ATOMIC_RELAXED, __HIP_MEMORY_SCOPE_AGENT);
  }
}

// --------------------------- out[dst] += alpha * h[src]   (wave per edge)
__global__ __launch_bounds__(256)
void k_edge_agg(const long long* __restrict__ ei,
                const float* __restrict__ a_s, const float* __restrict__ a_d,
                const float* __restrict__ mval, const float* __restrict__ den,
                const float* __restrict__ h, float* __restrict__ out,
                int E_, int EN) {
  long g = (long)blockIdx.x * blockDim.x + threadIdx.x;
  int edge = (int)(g >> 5);
  int lane = (int)(g & 31);
  if (edge >= EN) return;
  int s, d;
  if (edge < E_) { s = (int)ei[edge]; d = (int)ei[E_ + edge]; } else { s = d = edge - E_; }
  const float4 av = *(const float4*)(a_s + (size_t)s * 4);
  const float4 dv = *(const float4*)(a_d + (size_t)d * 4);
  const float4 mv = *(const float4*)(mval + (size_t)d * 4);
  const float4 dn = *(const float4*)(den + (size_t)d * 4);
  float alpha[4];
  alpha[0] = __expf(lrelu(av.x + dv.x) - mv.x) / dn.x;
  alpha[1] = __expf(lrelu(av.y + dv.y) - mv.y) / dn.y;
  alpha[2] = __expf(lrelu(av.z + dv.z) - mv.z) / dn.z;
  alpha[3] = __expf(lrelu(av.w + dv.w) - mv.w) / dn.w;
  const float* hs = h   + (size_t)s * 128 + lane;
  float*       od = out + (size_t)d * 128 + lane;
#pragma unroll
  for (int j = 0; j < 4; ++j) {
    __hip_atomic_fetch_add(od + j * 32, hs[j * 32] * alpha[j],
                           __ATOMIC_RELAXED, __HIP_MEMORY_SCOPE_AGENT);
  }
}

// ------------------------------- bias + ELU in place, per-channel stats
__global__ __launch_bounds__(128)
void k_bias_elu_stats(float* __restrict__ out, const float* __restrict__ bias,
                      float* __restrict__ bnsum, float* __restrict__ bnsumsq,
                      int n) {
  int c = threadIdx.x;                // one channel per thread
  int r0 = blockIdx.x * 256;
  int r1 = r0 + 256; if (r1 > n) r1 = n;
  float b = bias[c];
  float s = 0.f, q = 0.f;
  for (int r = r0; r < r1; ++r) {
    size_t idx = (size_t)r * 128 + c;
    float v = out[idx] + b;
    v = v > 0.f ? v : __expf(v) - 1.f;   // ELU(alpha=1)
    out[idx] = v;
    s += v; q += v * v;
  }
  __hip_atomic_fetch_add(&bnsum[c],   s, __ATOMIC_RELAXED, __HIP_MEMORY_SCOPE_AGENT);
  __hip_atomic_fetch_add(&bnsumsq[c], q, __ATOMIC_RELAXED, __HIP_MEMORY_SCOPE_AGENT);
}

// -------------------------------------------- BN coefficients (128 threads)
__global__ void k_bn_coef(const float* __restrict__ bnsum,
                          const float* __restrict__ bnsumsq,
                          const float* __restrict__ gamma,
                          const float* __restrict__ beta,
                          float* __restrict__ scale, float* __restrict__ shift,
                          int n) {
  int c = threadIdx.x;
  float inv_n = 1.0f / (float)n;
  float mean = bnsum[c] * inv_n;
  float var  = bnsumsq[c] * inv_n - mean * mean;
  float sc = gamma[c] * rsqrtf(var + 1e-5f);
  scale[c] = sc;
  shift[c] = beta[c] - mean * sc;
}

__global__ void k_bn_apply(float* __restrict__ out,
                           const float* __restrict__ scale,
                           const float* __restrict__ shift, long total) {
  long i = (long)blockIdx.x * blockDim.x + threadIdx.x;
  if (i < total) {
    int c = (int)(i & 127);
    out[i] = out[i] * scale[c] + shift[c];
  }
}

// ---------------------------------------------------------------- launcher
extern "C" void kernel_launch(void* const* d_in, const int* in_sizes, int n_in,
                              void* d_out, int out_size, void* d_ws, size_t ws_size,
                              hipStream_t stream) {
  (void)n_in; (void)out_size; (void)ws_size;
  const float*     x    = (const float*)d_in[0];
  const long long* ei   = (const long long*)d_in[1];   // int64 [2,E] flat
  const float*     W    = (const float*)d_in[2];
  const float*     asrc = (const float*)d_in[3];
  const float*     adst = (const float*)d_in[4];
  const float*     bias = (const float*)d_in[5];
  const float*     gamma= (const float*)d_in[6];
  const float*     beta = (const float*)d_in[7];
  float*           out  = (float*)d_out;

  const int N  = in_sizes[0] / 128;
  const int E  = in_sizes[1] / 2;
  const int EN = E + N;

  // workspace layout (all f32; 16B-aligned offsets)
  float* h     = (float*)d_ws;                 // N*128
  float* a_s   = h    + (size_t)N * 128;       // N*4
  float* a_d   = a_s  + (size_t)N * 4;         // N*4
  float* mbuf  = a_d  + (size_t)N * 4;         // N*4 (uint keys, then floats)
  float* den   = mbuf + (size_t)N * 4;         // N*4
  float* bnsum = den  + (size_t)N * 4;         // 128
  float* bnsq  = bnsum + 128;                  // 128
  float* scale = bnsq  + 128;                  // 128
  float* shift = scale + 128;                  // 128

  const long total = (long)N * 128;

  k_zero<<<(unsigned)((total + 255) / 256), 256, 0, stream>>>(out, total);
  k_zero<<<1, 256, 0, stream>>>(bnsum, 256);

  k_gemm<<<(unsigned)((N + 127) / 128), 256, 0, stream>>>(x, W, h, N);

  k_node<<<(unsigned)((N * 4 + 255) / 256), 256, 0, stream>>>(
      h, asrc, adst, a_s, a_d, (unsigned*)mbuf, den, N);

  k_edge_max<<<(unsigned)((EN + 255) / 256), 256, 0, stream>>>(
      ei, a_s, a_d, (unsigned*)mbuf, E, EN);

  k_mconv<<<(unsigned)((N * 4 + 255) / 256), 256, 0, stream>>>(
      (unsigned*)mbuf, N * 4);

  k_edge_sum<<<(unsigned)((EN + 255) / 256), 256, 0, stream>>>(
      ei, a_s, a_d, mbuf, den, E, EN);

  k_edge_agg<<<(unsigned)(((long)EN * 32 + 255) / 256), 256, 0, stream>>>(
      ei, a_s, a_d, mbuf, den, h, out, E, EN);

  k_bias_elu_stats<<<(unsigned)((N + 255) / 256), 128, 0, stream>>>(
      out, bias, bnsum, bnsq, N);

  k_bn_coef<<<1, 128, 0, stream>>>(bnsum, bnsq, gamma, beta, scale, shift, N);

  k_bn_apply<<<(unsigned)((total + 255) / 256), 256, 0, stream>>>(
      out, scale, shift, total);
}
